// GRUDecoderWithAttention_84533546320493
// MI455X (gfx1250) — compile-verified, matrix operand order checked by reading, not run on hardware
//
#include <hip/hip_runtime.h>
#include <hip/hip_bf16.h>

// ---------------------------------------------------------------------------
// GRU decoder with Bahdanau attention, teacher-forced, T=64 steps.
// B=32, T1=65, O=E=384, H=768. Sequential per-step chain, 6 kernels/step.
// GEMMs: v_wmma_f32_16x16x32_bf16; block = 4 N-tiles x 2 K-halves (split-K
// reduced through LDS, deterministic); one wave covers both 16-row M tiles
// so each weight row is fetched once. Wave index scalarized (readfirstlane);
// mainloop is a 2-stage pipeline whose halves are pinned with
// sched_barrier(0) so each WMMA pair always has the other buffer's 6 loads
// in flight behind its wait.
// ---------------------------------------------------------------------------

namespace {
constexpr int Bn  = 32;
constexpr int Tn  = 64;
constexpr int T1n = 65;
constexpr int On  = 384;
constexpr int En  = 384;
constexpr int Hn  = 768;
constexpr int H3n = 2304;   // 3*H
constexpr int CATn = 1536;  // H + O + E
constexpr int ATn  = 2080;  // T*(T+1)/2 attention slots per batch row
}

typedef __attribute__((ext_vector_type(16))) __bf16        v16bf;
typedef __attribute__((ext_vector_type(8)))  float         v8f;
typedef __attribute__((ext_vector_type(8)))  unsigned int  v8u;

__device__ __forceinline__ unsigned short f2bf(float f) {
  unsigned int u = __builtin_bit_cast(unsigned int, f);
  u += 0x7FFFu + ((u >> 16) & 1u);   // round-to-nearest-even
  return (unsigned short)(u >> 16);
}

__device__ __forceinline__ void load_chunk(
    const unsigned short* __restrict__ Ar0,
    const unsigned short* __restrict__ Ar1,
    const unsigned short* __restrict__ Wr,
    int k0, int kb, v8u& a0u, v8u& a1u, v8u& bu)
{
#pragma unroll
  for (int p = 0; p < 4; ++p) {
    const int k = k0 + kb + 2 * p;
    a0u[p]     = *(const unsigned int*)(Ar0 + k);
    a0u[p + 4] = *(const unsigned int*)(Ar0 + k + 16);
    a1u[p]     = *(const unsigned int*)(Ar1 + k);
    a1u[p + 4] = *(const unsigned int*)(Ar1 + k + 16);
    bu[p]      = *(const unsigned int*)(Wr + k);
    bu[p + 4]  = *(const unsigned int*)(Wr + k + 16);
  }
}

__device__ __forceinline__ void wmma_pair(
    const v8u& a0u, const v8u& a1u, const v8u& bu, v8f& acc0, v8f& acc1)
{
  v16bf a0 = __builtin_bit_cast(v16bf, a0u);
  v16bf a1 = __builtin_bit_cast(v16bf, a1u);
  v16bf b  = __builtin_bit_cast(v16bf, bu);
  acc0 = __builtin_amdgcn_wmma_f32_16x16x32_bf16(
      false, a0, false, b, (short)0, acc0, false, false);
  acc1 = __builtin_amdgcn_wmma_f32_16x16x32_bf16(
      false, a1, false, b, (short)0, acc1, false, false);
}

// Accumulate one K-range: acc0 += A[0:16] x W^T tile, acc1 += A[16:32] x W^T.
// Aoff already points at this range's first K column of row 0.
// 2-stage pipeline with sched_barrier(0) fences so the compiler can neither
// merge the two halves back into one chunk nor sink loads to their consumers.
__device__ __forceinline__ void seg_accum(
    const unsigned short* __restrict__ Aoff, int lda,
    const unsigned short* __restrict__ Wseg, int K,
    int kb, int lmod, v8f& acc0, v8f& acc1)
{
  const unsigned short* Ar0 = Aoff + (size_t)lmod * lda;
  const unsigned short* Ar1 = Aoff + (size_t)(16 + lmod) * lda;

  if ((K & 63) == 0) {          // all call sites: K in {192, 384, 768}
    v8u xa0, xa1, xb, ya0, ya1, yb;
    load_chunk(Ar0, Ar1, Wseg, 0, kb, xa0, xa1, xb);
    load_chunk(Ar0, Ar1, Wseg, 32, kb, ya0, ya1, yb);
    __builtin_amdgcn_sched_barrier(0);
    for (int k0 = 64; k0 < K; k0 += 64) {
      __builtin_prefetch(Wseg + k0 + 64, 0, 3);
      wmma_pair(xa0, xa1, xb, acc0, acc1);          // consumes prev X
      load_chunk(Ar0, Ar1, Wseg, k0, kb, xa0, xa1, xb);
      __builtin_amdgcn_sched_barrier(0);
      wmma_pair(ya0, ya1, yb, acc0, acc1);          // consumes prev Y
      load_chunk(Ar0, Ar1, Wseg, k0 + 32, kb, ya0, ya1, yb);
      __builtin_amdgcn_sched_barrier(0);
    }
    wmma_pair(xa0, xa1, xb, acc0, acc1);
    wmma_pair(ya0, ya1, yb, acc0, acc1);
  } else {                      // generic fallback (unused for current shapes)
    for (int k0 = 0; k0 < K; k0 += 32) {
      v8u a0u, a1u, bu;
      load_chunk(Ar0, Ar1, Wseg, k0, kb, a0u, a1u, bu);
      wmma_pair(a0u, a1u, bu, acc0, acc1);
    }
  }
}

// out[m][n] = sum_k [A0|A1|A2][m][k] * W[n][k] + bias[n]; M=32 fixed.
// Block: 8 waves = 4 consecutive N tiles x 2 K halves; halves reduced in LDS.
// N-range split (ntSplit) routes tiles to {outA,biasA,ldoA} or B-set, so two
// GEMMs sharing the same A can be fused via row-concatenated W.
// Optional epilogues (range A only): out2 (f32 strided), outbf (bf16 dense).
__global__ __launch_bounds__(256) void gemm_bf16_kernel(
    const unsigned short* __restrict__ A0, int lda0, int K0,
    const unsigned short* __restrict__ A1, int lda1, int K1,
    const unsigned short* __restrict__ A2, int lda2, int K2,
    const unsigned short* __restrict__ W, int ldw, int ntSplit,
    const float* __restrict__ biasA, float* __restrict__ outA, int ldoA,
    const float* __restrict__ biasB, float* __restrict__ outB, int ldoB,
    float* __restrict__ out2, size_t ld2m, int ld2n,
    unsigned short* __restrict__ outbf, int ldbf)
{
  const int tid   = threadIdx.x;
  // Scalarize the wave index: khalf/nt and all K-loop bounds become SGPRs,
  // keeping the WMMA loops scalar-counted and full-EXEC.
  const int wave  = __builtin_amdgcn_readfirstlane(tid >> 5);
  const int lane  = tid & 31;
  const int nt    = blockIdx.x * 4 + (wave >> 1);
  const int khalf = wave & 1;
  const int lhalf = lane >> 4;
  const int lmod  = lane & 15;
  const int kb = lhalf * 8;   // ISA 16-bit A/B layout: upper lane half K-offset

  const unsigned short* Wrow = W + (size_t)(nt * 16 + lmod) * ldw;

  const int Khalf = (K0 + K1 + K2) >> 1;
  const int klo = khalf * Khalf;
  const int khi = klo + Khalf;

  v8f acc0 = {}, acc1 = {};
  int base = 0;
  {
    const int lo = klo > base ? klo : base;
    const int hi = khi < base + K0 ? khi : base + K0;
    if (lo < hi) seg_accum(A0 + (lo - base), lda0, Wrow + lo, hi - lo,
                           kb, lmod, acc0, acc1);
    base += K0;
  }
  if (A1) {
    const int lo = klo > base ? klo : base;
    const int hi = khi < base + K1 ? khi : base + K1;
    if (lo < hi) seg_accum(A1 + (lo - base), lda1, Wrow + lo, hi - lo,
                           kb, lmod, acc0, acc1);
    base += K1;
  }
  if (A2) {
    const int lo = klo > base ? klo : base;
    const int hi = khi < base + K2 ? khi : base + K2;
    if (lo < hi) seg_accum(A2 + (lo - base), lda2, Wrow + lo, hi - lo,
                           kb, lmod, acc0, acc1);
  }

  // ---- deterministic split-K reduction through LDS ----
  __shared__ float red[4][32][16];
  if (khalf) {
#pragma unroll
    for (int r = 0; r < 8; ++r) {
      red[wave >> 1][lane][r]     = acc0[r];
      red[wave >> 1][lane][r + 8] = acc1[r];
    }
  }
  __syncthreads();
  if (khalf) return;
#pragma unroll
  for (int r = 0; r < 8; ++r) {
    acc0[r] += red[wave >> 1][lane][r];
    acc1[r] += red[wave >> 1][lane][r + 8];
  }

  // ---- epilogue (scalar range select) ----
  const bool rangeB = nt >= ntSplit;
  float* out        = rangeB ? outB  : outA;
  const float* bias = rangeB ? biasB : biasA;
  const int ldo     = rangeB ? ldoB  : ldoA;
  const int n  = (rangeB ? nt - ntSplit : nt) * 16 + lmod;
  const int m0 = lhalf * 8;               // C layout: m = r + 8*(lane>=16)
  const float bn = bias ? bias[n] : 0.0f;
#pragma unroll
  for (int r = 0; r < 8; ++r) {
    const float v0 = acc0[r] + bn;        // row m0+r
    const float v1 = acc1[r] + bn;        // row 16+m0+r
    out[(size_t)(m0 + r) * ldo + n]      = v0;
    out[(size_t)(16 + m0 + r) * ldo + n] = v1;
    if (out2) {
      out2[(size_t)(m0 + r) * ld2m + (size_t)n * ld2n]      = v0;
      out2[(size_t)(16 + m0 + r) * ld2m + (size_t)n * ld2n] = v1;
    }
    if (outbf) {
      outbf[(size_t)(m0 + r) * ldbf + n]      = f2bf(v0);
      outbf[(size_t)(16 + m0 + r) * ldbf + n] = f2bf(v1);
    }
  }
}

// ---------------------------------------------------------------------------
// One-time prep kernels
// ---------------------------------------------------------------------------
__global__ __launch_bounds__(256) void convert_bf16_kernel(
    const float* __restrict__ src, int srcLd, int colOff,
    unsigned short* __restrict__ dst, int cols)
{
  const int idx = blockIdx.x * 256 + threadIdx.x;
  const int r = idx / cols, c = idx % cols;
  dst[(size_t)r * cols + c] = f2bf(src[(size_t)r * srcLd + colOff + c]);
}

// Gather embeddings directly to bf16 for all (b, t<T).
__global__ __launch_bounds__(256) void embed_kernel(
    const int* __restrict__ target, const float* __restrict__ emb_table,
    unsigned short* __restrict__ emb_bf)
{
  const int idx = blockIdx.x * 256 + threadIdx.x;       // B*T*E
  const int b = idx / (Tn * En);
  const int rem = idx - b * Tn * En;
  const int t = rem / En;
  const int e = rem - t * En;
  const int tok = target[b * T1n + t];
  emb_bf[idx] = f2bf(emb_table[(size_t)tok * En + e]);
}

// h = hidden (f32 + bf16); zero hist row 0 and histproj row 0.
__global__ __launch_bounds__(256) void init_kernel(
    const float* __restrict__ hidden, float* __restrict__ h,
    unsigned short* __restrict__ hbf,
    float* __restrict__ hist, float* __restrict__ histproj)
{
  const int idx = blockIdx.x * 256 + threadIdx.x;       // B*H
  const int b = idx / Hn, j = idx - b * Hn;
  const float v = hidden[idx];
  h[idx]   = v;
  hbf[idx] = f2bf(v);
  histproj[((size_t)b * T1n) * Hn + j] = 0.0f;
  if (j < On) hist[((size_t)b * T1n) * On + j] = 0.0f;
}

// ---------------------------------------------------------------------------
// Per-step kernels
// ---------------------------------------------------------------------------
// Attention: one block per batch row; one wave per history position (strided),
// intra-wave shuffle reduction; softmax; context written straight to bf16.
__global__ __launch_bounds__(256) void attn_kernel(
    const float* __restrict__ hproj, const float* __restrict__ histproj,
    const float* __restrict__ hist,  const float* __restrict__ vW,
    unsigned short* __restrict__ wbf, float* __restrict__ attn_out, int i)
{
  const int b = blockIdx.x, tid = threadIdx.x;
  const int wave = __builtin_amdgcn_readfirstlane(tid >> 5);
  const int lane = tid & 31;
  __shared__ float sc[T1n];
  const float* hp = hproj + (size_t)b * Hn;

  for (int t = wave; t <= i; t += 8) {
    const float* hs = histproj + ((size_t)b * T1n + t) * Hn;
    float p = 0.0f;
    for (int k = lane; k < Hn; k += 32)
      p += vW[k] * tanhf(hp[k] + hs[k]);
#pragma unroll
    for (int s = 16; s > 0; s >>= 1) p += __shfl_down(p, s, 32);
    if (lane == 0) sc[t] = p;
  }
  __syncthreads();
  if (tid == 0) {
    float mx = sc[0];
    for (int t = 1; t <= i; ++t) mx = fmaxf(mx, sc[t]);
    float sum = 0.0f;
    for (int t = 0; t <= i; ++t) { float e = __expf(sc[t] - mx); sc[t] = e; sum += e; }
    const float inv = 1.0f / sum;
    for (int t = 0; t <= i; ++t) sc[t] *= inv;
  }
  __syncthreads();

  const int off = i * (i + 1) / 2;
  for (int t = tid; t <= i; t += 256)
    attn_out[(size_t)b * ATn + off + t] = sc[t];

  for (int o = tid; o < On; o += 256) {
    float acc = 0.0f;
    for (int t = 0; t <= i; ++t)
      acc += sc[t] * hist[((size_t)b * T1n + t) * On + o];
    wbf[(size_t)b * On + o] = f2bf(acc);
  }
}

// GRU gates; h updated in place, bf16 mirror refreshed.
__global__ __launch_bounds__(256) void gates_kernel(
    const float* __restrict__ gi, const float* __restrict__ gh,
    float* __restrict__ h, unsigned short* __restrict__ hbf)
{
  const int idx = blockIdx.x * 256 + threadIdx.x;       // B*H
  const int b = idx / Hn, j = idx - b * Hn;
  const float* gib = gi + (size_t)b * H3n;
  const float* ghb = gh + (size_t)b * H3n;
  const float r = 1.0f / (1.0f + __expf(-(gib[j] + ghb[j])));
  const float z = 1.0f / (1.0f + __expf(-(gib[Hn + j] + ghb[Hn + j])));
  const float n = tanhf(gib[2 * Hn + j] + r * ghb[2 * Hn + j]);
  const float hv = (1.0f - z) * n + z * h[idx];
  h[idx]   = hv;
  hbf[idx] = f2bf(hv);
}

// ---------------------------------------------------------------------------
extern "C" void kernel_launch(void* const* d_in, const int* in_sizes, int n_in,
                              void* d_out, int out_size, void* d_ws, size_t ws_size,
                              hipStream_t stream)
{
  (void)in_sizes; (void)n_in; (void)out_size; (void)ws_size;

  const int*   target    = (const int*)  d_in[0];
  const float* hidden    = (const float*)d_in[1];
  // d_in[2] = teacher_forcing_ratio (always 1, unused)
  const float* emb_table = (const float*)d_in[3];
  const float* attn_W    = (const float*)d_in[4];   // (H, H+O)
  const float* attn_b    = (const float*)d_in[5];   // (H,)
  const float* v_W       = (const float*)d_in[6];   // (1, H)
  const float* gru_Wih   = (const float*)d_in[7];   // (3H, E+O)
  const float* gru_Whh   = (const float*)d_in[8];   // (3H, H)
  const float* gru_bih   = (const float*)d_in[9];
  const float* gru_bhh   = (const float*)d_in[10];
  const float* fc_W      = (const float*)d_in[11];  // (O, H+O+E)
  const float* fc_b      = (const float*)d_in[12];

  float* outputs  = (float*)d_out;                              // B*O*T
  float* attn_out = (float*)d_out + (size_t)Bn * On * Tn;       // B*ATn

  // ---- workspace carve (256B aligned) ----
  size_t off = 0;
  char* base = (char*)d_ws;
  auto alloc = [&](size_t bytes) -> char* {
    char* p = base + off;
    off += (bytes + 255) & ~(size_t)255;
    return p;
  };
  float* h        = (float*)alloc(sizeof(float) * Bn * Hn);
  float* hproj    = (float*)alloc(sizeof(float) * Bn * Hn);
  float* gi       = (float*)alloc(sizeof(float) * Bn * H3n);
  float* gh       = (float*)alloc(sizeof(float) * Bn * H3n);
  float* hist     = (float*)alloc(sizeof(float) * Bn * T1n * On);
  float* histproj = (float*)alloc(sizeof(float) * Bn * T1n * Hn);
  // [Wh ; Whh] row-concatenated (N = H + 3H = 3072, K = 768) for fused GEMM.
  unsigned short* whhh_bf = (unsigned short*)alloc(sizeof(short) * (Hn + H3n) * Hn);
  unsigned short* wh_bf   = whhh_bf;                    // rows 0..H-1
  unsigned short* whh_bf  = whhh_bf + (size_t)Hn * Hn;  // rows H..H+3H-1
  unsigned short* wo_bf  = (unsigned short*)alloc(sizeof(short) * Hn * On);
  unsigned short* wih_bf = (unsigned short*)alloc(sizeof(short) * H3n * (En + On));
  unsigned short* fcw_bf = (unsigned short*)alloc(sizeof(short) * On * CATn);
  unsigned short* emb_bf = (unsigned short*)alloc(sizeof(short) * Bn * Tn * En);
  unsigned short* hbf    = (unsigned short*)alloc(sizeof(short) * Bn * Hn);
  unsigned short* wbf    = (unsigned short*)alloc(sizeof(short) * Bn * On);
  unsigned short* orowbf = (unsigned short*)alloc(sizeof(short) * Bn * On);

  // ---- one-time prep ----
  embed_kernel<<<(Bn * Tn * En) / 256, 256, 0, stream>>>(target, emb_table, emb_bf);
  init_kernel<<<(Bn * Hn) / 256, 256, 0, stream>>>(hidden, h, hbf, hist, histproj);
  convert_bf16_kernel<<<(Hn * Hn) / 256, 256, 0, stream>>>(attn_W, Hn + On, 0,  wh_bf, Hn);
  convert_bf16_kernel<<<(H3n * Hn) / 256, 256, 0, stream>>>(gru_Whh, Hn, 0, whh_bf, Hn);
  convert_bf16_kernel<<<(Hn * On) / 256, 256, 0, stream>>>(attn_W, Hn + On, Hn, wo_bf, On);
  convert_bf16_kernel<<<(H3n * (En + On)) / 256, 256, 0, stream>>>(gru_Wih, En + On, 0, wih_bf, En + On);
  convert_bf16_kernel<<<(On * CATn) / 256, 256, 0, stream>>>(fc_W, CATn, 0, fcw_bf, CATn);

  const int NOSPLIT = 1 << 30;

  // ---- sequential decoder steps (6 launches each) ----
  for (int i = 0; i < Tn; ++i) {
    const unsigned short* emb_i = emb_bf + (size_t)i * En;   // row stride Tn*En

    // Fused: [hproj | gh] = h @ [Wh;Whh]^T + [attn_b;gru_bhh]  (N=3072, K=768)
    gemm_bf16_kernel<<<((Hn + H3n) / 16) / 4, 256, 0, stream>>>(
        hbf, Hn, Hn, nullptr, 0, 0, nullptr, 0, 0,
        whhh_bf, Hn, /*ntSplit=*/Hn / 16,
        attn_b, hproj, Hn,
        gru_bhh, gh, H3n,
        nullptr, 0, 0, nullptr, 0);

    attn_kernel<<<Bn, 256, 0, stream>>>(hproj, histproj, hist, v_W,
                                        wbf, attn_out, i);

    // gi = [emb_i | w] @ Wih^T + bih   (N=2304, K=384+384)
    gemm_bf16_kernel<<<(H3n / 16) / 4, 256, 0, stream>>>(
        emb_i, Tn * En, En, wbf, On, On, nullptr, 0, 0,
        wih_bf, En + On, NOSPLIT,
        gru_bih, gi, H3n,
        nullptr, nullptr, 0,
        nullptr, 0, 0, nullptr, 0);

    gates_kernel<<<(Bn * Hn) / 256, 256, 0, stream>>>(gi, gh, h, hbf);

    // hist[:, i+1] = [h_new | w | emb_i] @ fc_W^T + fc_b   (N=384, K=1536)
    // epilogue also writes outputs[:, :, i] and bf16 row for histproj GEMM.
    gemm_bf16_kernel<<<(On / 16) / 4, 256, 0, stream>>>(
        hbf, Hn, Hn, wbf, On, On, emb_i, Tn * En, En,
        fcw_bf, CATn, NOSPLIT,
        fc_b, hist + (size_t)(i + 1) * On, T1n * On,
        nullptr, nullptr, 0,
        outputs + i, (size_t)On * Tn, Tn,
        orowbf, On);

    // histproj[:, i+1] = out @ Wo^T   (N=768, K=384)
    gemm_bf16_kernel<<<(Hn / 16) / 4, 256, 0, stream>>>(
        orowbf, On, On, nullptr, 0, 0, nullptr, 0, 0,
        wo_bf, On, NOSPLIT,
        nullptr, histproj + (size_t)(i + 1) * Hn, T1n * Hn,
        nullptr, nullptr, 0,
        nullptr, 0, 0, nullptr, 0);
  }
}
